// NPObsPrior_39075612459501
// MI455X (gfx1250) — compile-verified, matrix operand order checked by reading, not run on hardware
//
#include <hip/hip_runtime.h>
#include <math.h>

typedef __attribute__((ext_vector_type(16))) _Float16 v16h;
typedef __attribute__((ext_vector_type(8)))  _Float16 v8h;
typedef __attribute__((ext_vector_type(8)))  float    v8f;
typedef __attribute__((ext_vector_type(4)))  unsigned u32x4;

#define WAVES_PER_BLOCK 8
#define BLOCKS_PER_D    128
#define BATCH           131072
#define DDIM            8
#define HDIM            64
#define TILES           (BATCH / 16)
#define SLOPE           0.2f

__device__ __forceinline__ v8f wmma_f32out(v16h a, v16h b, v8f c) {
  return __builtin_amdgcn_wmma_f32_16x16x32_f16(
      false, a, false, b, (short)0, c, false, false);
}

// f16-accumulating WMMA: D/C are 16x16 f16 (4 VGPRs, half j <-> M = j + 8*hi).
// This layout is exactly the per-lane B-operand arrangement for the next layer.
__device__ __forceinline__ v8h wmma_f16out(v16h a, v16h b, v8h c) {
#if __has_builtin(__builtin_amdgcn_wmma_f16_16x16x32_f16)
  return __builtin_amdgcn_wmma_f16_16x16x32_f16(
      false, a, false, b, (short)0, c, false, false);
#else
  v8f cf = __builtin_convertvector(c, v8f);
  cf = wmma_f32out(a, b, cf);
  return __builtin_convertvector(cf, v8h);
#endif
}

__device__ __forceinline__ v8h splat8(float f) {
  v8h v;
  _Float16 x = (_Float16)f;
  #pragma unroll
  for (int j = 0; j < 8; ++j) v[j] = x;
  return v;
}

// per-f16-half sign mask: 0xFFFF where half < 0  (pinned to v_pk_ashrrev_i16)
__device__ __forceinline__ u32x4 mask8(v8h x) {
  u32x4 s = __builtin_bit_cast(u32x4, x);
  u32x4 r;
  const unsigned fifteen = 0x000F000F;   // shift 15 for both halves
  #pragma unroll
  for (int j = 0; j < 4; ++j) {
    unsigned m;
    asm("v_pk_ashrrev_i16 %0, %1, %2" : "=v"(m) : "v"(fifteen), "v"(s[j]));
    r[j] = m;
  }
  return r;
}

// per-half select: m ? a : b  (pinned to v_bfi_b32: (a&m)|(b&~m))
__device__ __forceinline__ v8h bfi8(u32x4 m, v8h a, v8h b) {
  u32x4 ua = __builtin_bit_cast(u32x4, a);
  u32x4 ub = __builtin_bit_cast(u32x4, b);
  u32x4 r;
  #pragma unroll
  for (int j = 0; j < 4; ++j) {
    unsigned o;
    asm("v_bfi_b32 %0, %1, %2, %3" : "=v"(o) : "v"(m[j]), "v"(ua[j]), "v"(ub[j]));
    r[j] = o;
  }
  return __builtin_bit_cast(v8h, r);
}

__device__ __forceinline__ v16h combine(v8h a, v8h b) {
  return __builtin_shufflevector(a, b, 0, 1, 2, 3, 4, 5, 6, 7,
                                        8, 9, 10, 11, 12, 13, 14, 15);
}

__global__ __launch_bounds__(256, 4) void npobs_mlp_kernel(
    const float* __restrict__ z,   const float* __restrict__ xs,
    const float* __restrict__ W0,  const float* __restrict__ b0,
    const float* __restrict__ W1,  const float* __restrict__ b1,
    const float* __restrict__ W2,  const float* __restrict__ b2,
    const float* __restrict__ Wout,const float* __restrict__ bout,
    float* __restrict__ out, float* __restrict__ partial)
{
  const int d    = blockIdx.y;
  const int lane = threadIdx.x & 31;
  const int wid  = threadIdx.x >> 5;
  const int lo   = lane & 15;
  const int hi   = lane >> 4;

  const float* W0d = W0 + d * HDIM * 9;
  const float* b0d = b0 + d * HDIM;
  const float* W1d = W1 + d * HDIM * HDIM;
  const float* b1d = b1 + d * HDIM;
  const float* W2d = W2 + d * HDIM * HDIM;
  const float* b2d = b2 + d * HDIM;
  const float* Wod = Wout + d * HDIM;
  const float  boutv = bout[d];

  // ---- per-lane pre-swizzled operands in LDS (b128 fetches) ----
  __shared__ __align__(32) v16h a0l[4][32];     // layer0 A-tiles (bias @ k=9)
  __shared__ __align__(32) v16h a2l[4][2][32];  // layer2 A-tiles (W2)
  __shared__ __align__(16) v8h  b1l[4][32];     // bias1, 16-bit C-layout (f16)
  __shared__ __align__(16) v8h  w0ll[4][32];    // W0[:, -1], 16-bit C-layout (f16)
  __shared__ __align__(16) float b2l[64];       // bias2 (f32 C-init)
  __shared__ __align__(16) float wol[64];       // Wout  (f32 final dot)

  for (int idx = threadIdx.x; idx < 4 * 32; idx += 256) {
    const int m  = idx >> 5;
    const int ln = idx & 31;
    const int llo = ln & 15, lhi = ln >> 4;
    const int g  = 16 * m + llo;       // A-tile row
    const int g0 = 16 * m + 8 * lhi;   // C-layout base (half j -> g0 + j)
    v16h u;
    #pragma unroll
    for (int j = 0; j < 16; ++j) u[j] = (_Float16)0.0f;
    if (lhi == 0) {
      #pragma unroll
      for (int j = 0; j < 8; ++j) u[j] = (_Float16)W0d[g * 9 + j];  // k=0..7
    } else {
      u[0] = (_Float16)W0d[g * 9 + 8];  // k=8 : xs coefficient
      u[1] = (_Float16)b0d[g];          // k=9 : bias column (input supplies 1)
    }
    a0l[m][ln] = u;
    v8h pb, pw;
    #pragma unroll
    for (int k = 0; k < 8; ++k) {
      pb[k] = (_Float16)b1d[g0 + k];
      pw[k] = (_Float16)W0d[(g0 + k) * 9 + 8];
    }
    b1l[m][ln]  = pb;
    w0ll[m][ln] = pw;
  }
  // layer-2 weight A-tiles into LDS (4 m-tiles x 2 k-tiles x 32 lanes)
  for (int idx = threadIdx.x; idx < 4 * 2 * 32; idx += 256) {
    const int m  = idx >> 6;
    const int k  = (idx >> 5) & 1;
    const int ln = idx & 31;
    const int llo = ln & 15, lhi = ln >> 4;
    const float* p2 = W2d + (16 * m + llo) * HDIM + 32 * k + 8 * lhi;
    v16h u;
    #pragma unroll
    for (int j = 0; j < 8; ++j) {
      u[j]     = (_Float16)p2[j];
      u[8 + j] = (_Float16)p2[16 + j];
    }
    a2l[m][k][ln] = u;
  }
  for (int idx = threadIdx.x; idx < HDIM; idx += 256) {
    b2l[idx] = b2d[idx];
    wol[idx] = Wod[idx];
  }
  __syncthreads();

  // ---- register-resident W1 A-tiles only (64 VGPRs) ----
  v16h a1[4][2];
  #pragma unroll
  for (int m = 0; m < 4; ++m) {
    const int g = 16 * m + lo;
    #pragma unroll
    for (int k = 0; k < 2; ++k) {
      v16h u1;
      const float* p1 = W1d + g * HDIM + 32 * k + 8 * hi;
      #pragma unroll
      for (int j = 0; j < 8; ++j) {
        u1[j]     = (_Float16)p1[j];
        u1[8 + j] = (_Float16)p1[16 + j];
      }
      a1[m][k] = u1;
    }
  }

  const v8h S8 = splat8(SLOPE);
  float lacc = 0.0f;
  const int gw = blockIdx.x * WAVES_PER_BLOCK + wid;

  for (int t = gw; t < TILES; t += BLOCKS_PER_D * WAVES_PER_BLOCK) {
    // Launder lane so the per-lane LDS loads stay in-loop (no LICM -> spill).
    int llane = lane;
    asm volatile("" : "+v"(llane));
    const int lhi2 = (llane >> 4);

    const int b = t * 16 + lo;

    // ---- input B-tile: k=0..7 -> z, k=8 -> xs, k=9 -> 1.0 ----
    v16h ub;
    #pragma unroll
    for (int j = 0; j < 16; ++j) ub[j] = (_Float16)0.0f;
    {
      const float* zr = z + (size_t)b * 8;
      const float  xv = xs[(size_t)b * 8 + d];
      if (hi == 0) {
        #pragma unroll
        for (int j = 0; j < 8; ++j) ub[j] = (_Float16)zr[j];
      } else {
        ub[0] = (_Float16)xv;
        ub[1] = (_Float16)1.0f;
      }
    }

    const v8h zero8 = splat8(0.0f);
    v16h hB[2], tB[2];

    // ---- layer 0 : 4 f16-out WMMA; D tiles feed B operands directly ----
    {
      v8h c0[4];
      #pragma unroll
      for (int m = 0; m < 4; ++m)
        c0[m] = wmma_f16out(a0l[m][llane], ub, zero8);
      v8h nh[4], nt[4];
      #pragma unroll
      for (int m = 0; m < 4; ++m) {
        u32x4 ms = mask8(c0[m]);                 // pre < 0 per half
        nh[m] = bfi8(ms, c0[m] * S8, c0[m]);     // LeakyReLU
        v8h wt = w0ll[m][llane];
        nt[m] = bfi8(ms, wt * S8, wt);           // W0[:,-1] * dmask
      }
      hB[0] = combine(nh[0], nh[1]); hB[1] = combine(nh[2], nh[3]);
      tB[0] = combine(nt[0], nt[1]); tB[1] = combine(nt[2], nt[3]);
    }

    // ---- layer 1 : 16 f16-out WMMA in pairs; bias1 rides in C operand ----
    {
      v8h nh[4], nt[4];
      #pragma unroll
      for (int mp = 0; mp < 2; ++mp) {
        v8h cH[2], cT[2];
        #pragma unroll
        for (int q = 0; q < 2; ++q) {
          const int m = 2 * mp + q;
          v8h x = wmma_f16out(a1[m][0], hB[0], b1l[m][llane]);
          cH[q]   = wmma_f16out(a1[m][1], hB[1], x);
          v8h y = wmma_f16out(a1[m][0], tB[0], zero8);
          cT[q]   = wmma_f16out(a1[m][1], tB[1], y);
        }
        #pragma unroll
        for (int q = 0; q < 2; ++q) {
          const int m = 2 * mp + q;
          u32x4 ms = mask8(cH[q]);
          nh[m] = bfi8(ms, cH[q] * S8, cH[q]);
          nt[m] = bfi8(ms, cT[q] * S8, cT[q]);
        }
      }
      hB[0] = combine(nh[0], nh[1]); hB[1] = combine(nh[2], nh[3]);
      tB[0] = combine(nt[0], nt[1]); tB[1] = combine(nt[2], nt[3]);
    }

    // ---- layer 2 : 16 f32-out WMMA in pairs, fused with output dot ----
    {
      float res = 0.0f, dout = 0.0f;
      #pragma unroll
      for (int mp = 0; mp < 2; ++mp) {
        v16h a2t[2][2];
        #pragma unroll
        for (int q = 0; q < 2; ++q) {
          a2t[q][0] = a2l[2 * mp + q][0][llane];
          a2t[q][1] = a2l[2 * mp + q][1][llane];
        }
        v8f cH[2], cT[2];
        #pragma unroll
        for (int q = 0; q < 2; ++q) {
          const int m = 2 * mp + q;
          const float* bp = &b2l[16 * m + 8 * lhi2];
          v8f x;
          #pragma unroll
          for (int r = 0; r < 8; ++r) x[r] = bp[r];   // bias2 via accumulator
          x = wmma_f32out(a2t[q][0], hB[0], x);
          cH[q] = wmma_f32out(a2t[q][1], hB[1], x);
          v8f y = {};
          y = wmma_f32out(a2t[q][0], tB[0], y);
          cT[q] = wmma_f32out(a2t[q][1], tB[1], y);
        }
        #pragma unroll
        for (int q = 0; q < 2; ++q) {
          const int m = 2 * mp + q;
          const float* wp = &wol[16 * m + 8 * lhi2];
          v8f wv;
          #pragma unroll
          for (int r = 0; r < 8; ++r) wv[r] = wp[r];
          #pragma unroll
          for (int r = 0; r < 8; ++r) {
            const float p  = cH[q][r];
            const float hv = fmaxf(p, SLOPE * p);
            const float tv = (p >= 0.0f) ? cT[q][r] : SLOPE * cT[q][r];
            res  = fmaf(hv, wv[r], res);
            dout = fmaf(tv, wv[r], dout);
          }
        }
      }
      res  += __shfl_xor(res,  16, 32);
      dout += __shfl_xor(dout, 16, 32);
      if (hi == 0) {
        out[(size_t)b * DDIM + d] = res + boutv;   // residuals[b, 0, d]
        lacc += logf(fabsf(dout));
      }
    }
  }

  // ---- deterministic block partial for sum(log|dout|) ----
  #pragma unroll
  for (int s = 1; s < 32; s <<= 1) lacc += __shfl_xor(lacc, s, 32);
  __shared__ float sred[WAVES_PER_BLOCK];
  if (lane == 0) sred[wid] = lacc;
  __syncthreads();
  if (threadIdx.x == 0) {
    float s = 0.0f;
    #pragma unroll
    for (int i = 0; i < WAVES_PER_BLOCK; ++i) s += sred[i];
    partial[blockIdx.y * BLOCKS_PER_D + blockIdx.x] = s;
  }
}

__global__ void npobs_reduce_kernel(const float* __restrict__ partial,
                                    float* __restrict__ out, int n) {
  if (threadIdx.x == 0 && blockIdx.x == 0) {
    float s = 0.0f;
    for (int i = 0; i < n; ++i) s += partial[i];   // fixed order -> deterministic
    out[(size_t)BATCH * DDIM] = s;
  }
}

extern "C" void kernel_launch(void* const* d_in, const int* in_sizes, int n_in,
                              void* d_out, int out_size, void* d_ws, size_t ws_size,
                              hipStream_t stream) {
  (void)in_sizes; (void)n_in; (void)out_size; (void)ws_size;
  const float* z    = (const float*)d_in[0];
  const float* xs   = (const float*)d_in[1];
  const float* W0   = (const float*)d_in[2];
  const float* b0   = (const float*)d_in[3];
  const float* W1   = (const float*)d_in[4];
  const float* b1   = (const float*)d_in[5];
  const float* W2   = (const float*)d_in[6];
  const float* b2   = (const float*)d_in[7];
  const float* Wout = (const float*)d_in[8];
  const float* bout = (const float*)d_in[9];
  float* out     = (float*)d_out;
  float* partial = (float*)d_ws;   // BLOCKS_PER_D * DDIM floats

  dim3 grid(BLOCKS_PER_D, DDIM);
  npobs_mlp_kernel<<<grid, WAVES_PER_BLOCK * 32, 0, stream>>>(
      z, xs, W0, b0, W1, b1, W2, b2, Wout, bout, out, partial);
  npobs_reduce_kernel<<<1, 32, 0, stream>>>(partial, out, BLOCKS_PER_D * DDIM);
}
